// MLPKG_34162169872496
// MI455X (gfx1250) — matrix operand order repaired and checked by reading, not compiled, not run on hardware
//
#include <hip/hip_runtime.h>
#include <cstdint>

typedef __attribute__((ext_vector_type(16))) __bf16 v16bf;
typedef __attribute__((ext_vector_type(8)))  float  v8f;
typedef __attribute__((ext_vector_type(4)))  float  v4f;

#define N_NODES 65536

// ---------- f32 -> bf16 bulk weight conversion ----------
__global__ __launch_bounds__(256) void cvt_f32_bf16(const float* __restrict__ s,
                                                    __bf16* __restrict__ d, int n) {
  int i = blockIdx.x * 256 + threadIdx.x;
  if (i < n) d[i] = (__bf16)s[i];
}

// ---------- build transposed zero-padded bias: Bbt[c][k] = (k<9) ? b[k][c] : 0 ----------
__global__ __launch_bounds__(256) void build_bbt(const float* __restrict__ b,
                                                 __bf16* __restrict__ d, int O) {
  int i = blockIdx.x * 256 + threadIdx.x;  // over O*32
  int c = i >> 5;
  int k = i & 31;
  float v = (k < 9) ? b[k * O + c] : 0.0f;
  d[i] = (__bf16)v;
}

// ---------- fused relation-mixed GEMM layer ----------
// Out[n,c] = relu( sum_r e[n,r] * sum_k Act[n,k]*W[r,c,k]  +  sum_r e[n,r]*b[r,c] )
// Block: 256 threads (8 waves). Wave w -> rows [bx*128 + w*16, +16). Block cols: [by*128, +128).
template<int K, int O, bool RELU>
__global__ __launch_bounds__(256) void kg_layer(
    const float*  __restrict__ Act,   // [N][K]  f32
    const float*  __restrict__ Edge,  // [N][9]  f32
    const __bf16* __restrict__ Wb,    // [9][O][K] bf16
    const __bf16* __restrict__ Bbt,   // [O][32]   bf16 (zero padded, transposed)
    float*        __restrict__ Out)   // [N][O]  f32
{
  const int lane = threadIdx.x & 31;
  const int wave = threadIdx.x >> 5;
  const int ln   = lane & 15;
  const int hi   = lane >> 4;            // half-wave select
  const int m0   = blockIdx.x * 128 + wave * 16;
  const int cb   = blockIdx.y * 128;
  const int row  = m0 + ln;              // A-fragment: lane holds row (m = lane%16)
  const int klo  = hi ? 16 : 0;          // A-frag K offset (contiguous k = 16*hi + j)
  const int klo2 = hi ? 16 : 0;          // B-frag K offset (contiguous k = 16*hi + j)

  // per-lane relation weights for this node row
  float e9[9];
#pragma unroll
  for (int r = 0; r < 9; ++r) e9[r] = Edge[(size_t)row * 9 + r];

  // A-fragment of edge_attr for the bias WMMA (K' = 32, only k'<9 valid)
  // lanes 0-15 hold K'=0..15 (elems 0..8 valid), lanes 16-31 hold K'=16..31 -> all zero
  v16bf ef;
#pragma unroll
  for (int j = 0; j < 8; ++j)
    ef[j] = hi ? (__bf16)0.0f : (__bf16)e9[j];
#pragma unroll
  for (int j = 8; j < 16; ++j) ef[j] = (j == 8 && !hi) ? (__bf16)e9[8] : (__bf16)0.0f;

  v8f acc[8] = {};  // 8 column tiles of 16x16 f32

  const float*  arow = Act + (size_t)row * K + klo;
  const __bf16* wcol = Wb + (size_t)(cb + ln) * K + klo2;  // + r*O*K + ct*16*K later

#pragma unroll 2
  for (int kc = 0; kc < K / 32; ++kc) {
    const int kb = kc * 32;
    // A tile (unscaled, f32): lane needs k = kb+klo..+15
    v4f x0 = *(const v4f*)(arow + kb);
    v4f x1 = *(const v4f*)(arow + kb + 4);
    v4f x2 = *(const v4f*)(arow + kb + 8);
    v4f x3 = *(const v4f*)(arow + kb + 12);
    float ax[16];
#pragma unroll
    for (int j = 0; j < 4; ++j) {
      ax[j] = x0[j]; ax[4 + j] = x1[j]; ax[8 + j] = x2[j]; ax[12 + j] = x3[j];
    }

#pragma unroll
    for (int r = 0; r < 9; ++r) {
      // per-relation row scale folded into the A fragment (per-lane scalar)
      const float s = e9[r];
      v16bf af;
#pragma unroll
      for (int j = 0; j < 16; ++j) af[j] = (__bf16)(ax[j] * s);

      const __bf16* wr = wcol + ((size_t)r * O) * K + kb;
#pragma unroll
      for (int ct = 0; ct < 8; ++ct) {
        v16bf bfrag = *(const v16bf*)(wr + (size_t)ct * 16 * K);
        acc[ct] = __builtin_amdgcn_wmma_f32_16x16x32_bf16(
            false, af, false, bfrag, (short)0, acc[ct], false, false);
      }
    }
  }

  // bias: acc += ef(16x32, e padded) x Bbt(32x16)  -- one extra WMMA per tile
#pragma unroll
  for (int ct = 0; ct < 8; ++ct) {
    v16bf bb = *(const v16bf*)(Bbt + (size_t)(cb + ct * 16 + ln) * 32 + klo2);
    acc[ct] = __builtin_amdgcn_wmma_f32_16x16x32_bf16(
        false, ef, false, bb, (short)0, acc[ct], false, false);
  }

  // store: C/D layout -> lane(0-15): M=v, N=lane ; lane(16-31): M=8+v, N=lane-16
  const int r0 = m0 + (hi ? 8 : 0);
#pragma unroll
  for (int ct = 0; ct < 8; ++ct) {
    const int c = cb + ct * 16 + ln;
#pragma unroll
    for (int v = 0; v < 8; ++v) {
      float val = acc[ct][v];
      if (RELU) val = fmaxf(val, 0.0f);
      Out[(size_t)(r0 + v) * O + c] = val;
    }
  }
}

// ---------- workspace layout ----------
static constexpr size_t W0N = 9ull * 512 * 256;
static constexpr size_t W1N = 9ull * 512 * 512;
static constexpr size_t W2N = 9ull * 512 * 512;
static constexpr size_t W3N = 9ull * 256 * 512;
static constexpr size_t OFF_W0 = 0;
static constexpr size_t OFF_W1 = OFF_W0 + W0N * 2;
static constexpr size_t OFF_W2 = OFF_W1 + W1N * 2;
static constexpr size_t OFF_W3 = OFF_W2 + W2N * 2;
static constexpr size_t OFF_B0 = OFF_W3 + W3N * 2;
static constexpr size_t OFF_B1 = OFF_B0 + 512 * 32 * 2;
static constexpr size_t OFF_B2 = OFF_B1 + 512 * 32 * 2;
static constexpr size_t OFF_B3 = OFF_B2 + 512 * 32 * 2;
static constexpr size_t OFF_AA = (OFF_B3 + 256 * 32 * 2 + 255) & ~size_t(255);
static constexpr size_t OFF_AB = OFF_AA + (size_t)N_NODES * 512 * 4;

extern "C" void kernel_launch(void* const* d_in, const int* in_sizes, int n_in,
                              void* d_out, int out_size, void* d_ws, size_t ws_size,
                              hipStream_t stream) {
  const float* x  = (const float*)d_in[0];
  const float* e  = (const float*)d_in[1];
  const float* W0 = (const float*)d_in[2];
  const float* b0 = (const float*)d_in[3];
  const float* W1 = (const float*)d_in[4];
  const float* b1 = (const float*)d_in[5];
  const float* W2 = (const float*)d_in[6];
  const float* b2 = (const float*)d_in[7];
  const float* W3 = (const float*)d_in[8];
  const float* b3 = (const float*)d_in[9];

  char* ws = (char*)d_ws;
  __bf16* W0b = (__bf16*)(ws + OFF_W0);
  __bf16* W1b = (__bf16*)(ws + OFF_W1);
  __bf16* W2b = (__bf16*)(ws + OFF_W2);
  __bf16* W3b = (__bf16*)(ws + OFF_W3);
  __bf16* B0t = (__bf16*)(ws + OFF_B0);
  __bf16* B1t = (__bf16*)(ws + OFF_B1);
  __bf16* B2t = (__bf16*)(ws + OFF_B2);
  __bf16* B3t = (__bf16*)(ws + OFF_B3);
  float*  actA = (float*)(ws + OFF_AA);
  float*  actB = (float*)(ws + OFF_AB);

  // ---- pre-pass: weight/bias conversion (small; L2-resident afterwards) ----
  cvt_f32_bf16<<<(int)(W0N / 256), 256, 0, stream>>>(W0, W0b, (int)W0N);
  cvt_f32_bf16<<<(int)(W1N / 256), 256, 0, stream>>>(W1, W1b, (int)W1N);
  cvt_f32_bf16<<<(int)(W2N / 256), 256, 0, stream>>>(W2, W2b, (int)W2N);
  cvt_f32_bf16<<<(int)(W3N / 256), 256, 0, stream>>>(W3, W3b, (int)W3N);
  build_bbt<<<512 * 32 / 256, 256, 0, stream>>>(b0, B0t, 512);
  build_bbt<<<512 * 32 / 256, 256, 0, stream>>>(b1, B1t, 512);
  build_bbt<<<512 * 32 / 256, 256, 0, stream>>>(b2, B2t, 512);
  build_bbt<<<256 * 32 / 256, 256, 0, stream>>>(b3, B3t, 256);

  // ---- 4 fused layers ----
  dim3 blk(256);
  kg_layer<256, 512, true ><<<dim3(N_NODES / 128, 512 / 128), blk, 0, stream>>>(x,    e, W0b, B0t, actA);
  kg_layer<512, 512, true ><<<dim3(N_NODES / 128, 512 / 128), blk, 0, stream>>>(actA, e, W1b, B1t, actB);
  kg_layer<512, 512, true ><<<dim3(N_NODES / 128, 512 / 128), blk, 0, stream>>>(actB, e, W2b, B2t, actA);
  kg_layer<512, 256, false><<<dim3(N_NODES / 128, 256 / 128), blk, 0, stream>>>(actA, e, W3b, B3t, (float*)d_out);
}
